// MH_GCAT_53747220742803
// MI455X (gfx1250) — matrix operand order repaired and checked by reading, not compiled, not run on hardware
//
#include <hip/hip_runtime.h>

#define N_NODES 30000
#define N_EDGES 480000
#define F_IN    376
#define KP1     384     // F_IN padded to multiple of 32 for WMMA K-steps
#define HID     512
#define DF      1536
#define NEG_SLOPE 0.2f

typedef __attribute__((ext_vector_type(8)))  float  v8f;
typedef __attribute__((ext_vector_type(16))) __bf16 v16bf;
typedef __attribute__((ext_vector_type(8)))  __bf16 v8bf;

// ---------- helpers ----------
__device__ __forceinline__ unsigned short f2bf(float f) {
  unsigned u = __float_as_uint(f);
  u += 0x7FFFu + ((u >> 16) & 1u);          // round-to-nearest-even
  return (unsigned short)(u >> 16);
}
// order-preserving float<->uint map for atomicMax on floats
__device__ __forceinline__ unsigned f32_ord(float x) {
  unsigned u = __float_as_uint(x);
  return (u & 0x80000000u) ? ~u : (u | 0x80000000u);
}
__device__ __forceinline__ float ord_f32(unsigned u) {
  return __uint_as_float((u & 0x80000000u) ? (u & 0x7FFFFFFFu) : ~u);
}

// ---------- elementwise / setup kernels ----------
__global__ void fill_u32(unsigned* p, unsigned v, long n) {
  long i = (long)blockIdx.x * blockDim.x + threadIdx.x;
  if (i < n) p[i] = v;
}
__global__ void deg_k(const int* __restrict__ dst, const float* __restrict__ w,
                      float* __restrict__ deg) {
  int e = blockIdx.x * blockDim.x + threadIdx.x;
  if (e < N_EDGES) atomicAdd(deg + dst[e], w[e]);
}
__global__ void dinv_k(const float* __restrict__ deg, float* __restrict__ dinv) {
  int i = blockIdx.x * blockDim.x + threadIdx.x;
  if (i < N_NODES) dinv[i] = rsqrtf(deg[i] + 1.0f);   // +1 self loop; always >0
}
__global__ void norm_k(const int* __restrict__ s, const int* __restrict__ t,
                       const float* __restrict__ w, const float* __restrict__ dinv,
                       float* __restrict__ nw) {
  int e = blockIdx.x * blockDim.x + threadIdx.x;
  if (e < N_EDGES) nw[e] = dinv[s[e]] * w[e] * dinv[t[e]];
}
__global__ void conv_x(const float* __restrict__ x, unsigned short* __restrict__ xb) {
  int idx = blockIdx.x * blockDim.x + threadIdx.x;
  if (idx >= N_NODES * KP1) return;
  int i = idx / KP1, k = idx - i * KP1;
  xb[idx] = (k < F_IN) ? f2bf(x[(long)i * F_IN + k]) : (unsigned short)0;
}
__global__ void conv_wt(const float* __restrict__ W, unsigned short* __restrict__ Wt,
                        int K, int Kp, int Nout) {
  int idx = blockIdx.x * blockDim.x + threadIdx.x;
  if (idx >= Nout * Kp) return;
  int n = idx / Kp, k = idx - n * Kp;
  Wt[idx] = (k < K) ? f2bf(W[(long)k * Nout + n]) : (unsigned short)0;
}

// ---------- WMMA GEMM: C[M,Nout](f32) = A[M,K](bf16) * Bt[Nout,K](bf16)^T ----------
// one wave computes one 16x16 tile; K stepped by 32 via v_wmma_f32_16x16x32_bf16
__global__ void gemm_bf16_wmma(const unsigned short* __restrict__ A,
                               const unsigned short* __restrict__ Bt,
                               float* __restrict__ C,
                               int M, int Nout, int K, int acc) {
  const int lane = threadIdx.x & 31;
  const int wave = (blockIdx.x * blockDim.x + threadIdx.x) >> 5;
  const int ntn  = Nout >> 4;
  if (wave >= (M >> 4) * ntn) return;           // wave-uniform: EXEC stays all-1s
  const int tm = wave / ntn;
  const int tn = wave - tm * ntn;

  // A fragment: lane holds row (lane&15), 16 consecutive K from kb=(lane>>4)*16
  const long arow = (long)(tm * 16 + (lane & 15)) * K + ((lane >> 4) * 16);
  // B fragment: lane holds column (lane&15), 16 consecutive K from kb=(lane>>4)*16
  const long brow = (long)(tn * 16 + (lane & 15)) * K + ((lane >> 4) * 16);
  // C/D: lane holds col (lane&15), rows (lane>>4)*8 + r in vgpr r
  const int crow = tm * 16 + (lane >> 4) * 8;
  const int ccol = tn * 16 + (lane & 15);
  float* cp = C + (long)crow * Nout + ccol;

  v8f c = {};
  if (acc) {
#pragma unroll
    for (int r = 0; r < 8; ++r) c[r] = cp[(long)r * Nout];
  }
  for (int k0 = 0; k0 < K; k0 += 32) {
    v8bf a0 = *reinterpret_cast<const v8bf*>(A + arow + k0);
    v8bf a1 = *reinterpret_cast<const v8bf*>(A + arow + k0 + 8);
    v8bf b0 = *reinterpret_cast<const v8bf*>(Bt + brow + k0);
    v8bf b1 = *reinterpret_cast<const v8bf*>(Bt + brow + k0 + 8);
    v16bf av = __builtin_shufflevector(a0, a1, 0,1,2,3,4,5,6,7,8,9,10,11,12,13,14,15);
    v16bf bv = __builtin_shufflevector(b0, b1, 0,1,2,3,4,5,6,7,8,9,10,11,12,13,14,15);
    c = __builtin_amdgcn_wmma_f32_16x16x32_bf16(false, av, false, bv,
                                                (short)0, c, false, false);
  }
#pragma unroll
  for (int r = 0; r < 8; ++r) cp[(long)r * Nout] = c[r];
}

// ---------- sparse aggregation (scatter with f32 atomics) ----------
__global__ void gcn_agg(const int* __restrict__ src, const int* __restrict__ dst,
                        const float* __restrict__ normw, const float* __restrict__ h,
                        float* __restrict__ agg) {
  long idx = (long)blockIdx.x * blockDim.x + threadIdx.x;
  if (idx >= (long)N_EDGES * (HID / 4)) return;
  int e = (int)(idx >> 7);
  int c = ((int)idx & 127) * 4;
  float w = normw[e];
  const float4 v = *reinterpret_cast<const float4*>(h + (long)src[e] * HID + c);
  float* o = agg + (long)dst[e] * HID + c;
  atomicAdd(o + 0, w * v.x); atomicAdd(o + 1, w * v.y);
  atomicAdd(o + 2, w * v.z); atomicAdd(o + 3, w * v.w);
}
__global__ void sage_agg(const int* __restrict__ src, const int* __restrict__ dst,
                         const float* __restrict__ x1dense, float* __restrict__ sums) {
  long idx = (long)blockIdx.x * blockDim.x + threadIdx.x;
  if (idx >= (long)N_EDGES * (HID / 4)) return;
  int e = (int)(idx >> 7);
  int c = ((int)idx & 127) * 4;
  const float4 v = *reinterpret_cast<const float4*>(x1dense + (long)src[e] * DF + c);
  float* o = sums + (long)dst[e] * HID + c;
  atomicAdd(o + 0, v.x); atomicAdd(o + 1, v.y);
  atomicAdd(o + 2, v.z); atomicAdd(o + 3, v.w);
}
__global__ void cnt_k(const int* __restrict__ dst, float* __restrict__ cnt) {
  int e = blockIdx.x * blockDim.x + threadIdx.x;
  if (e < N_EDGES) atomicAdd(cnt + dst[e], 1.0f);
}

// ---------- epilogues (self-loop term + bias + relu; write strided dense out) ----------
__global__ void gcn_epi(const float* __restrict__ agg, const float* __restrict__ h,
                        const float* __restrict__ dinv, const float* __restrict__ bias,
                        unsigned short* __restrict__ xb16, float* __restrict__ dense,
                        int doff) {
  int idx = blockIdx.x * blockDim.x + threadIdx.x;
  if (idx >= N_NODES * HID) return;
  int i = idx >> 9, f = idx & 511;
  float di = dinv[i];
  float v = fmaxf(agg[idx] + di * di * h[idx] + bias[f], 0.0f);
  if (xb16) xb16[idx] = f2bf(v);
  dense[(long)i * DF + doff + f] = v;
}
__global__ void mean_k(const float* __restrict__ sums, const float* __restrict__ cnt,
                       unsigned short* __restrict__ mb) {
  int idx = blockIdx.x * blockDim.x + threadIdx.x;
  if (idx >= N_NODES * HID) return;
  mb[idx] = f2bf(sums[idx] / fmaxf(cnt[idx >> 9], 1.0f));
}
__global__ void sage_epi(const float* __restrict__ pre, const float* __restrict__ bias,
                         float* __restrict__ dense) {
  int idx = blockIdx.x * blockDim.x + threadIdx.x;
  if (idx >= N_NODES * HID) return;
  int i = idx >> 9, f = idx & 511;
  dense[(long)i * DF + 1024 + f] = fmaxf(pre[idx] + bias[f], 0.0f);
}

// ---------- GAT ----------
template <int H>
__global__ void gat_proj(const float* __restrict__ dense, const float* __restrict__ W,
                         float* __restrict__ hout) {
  extern __shared__ float sW[];                  // 1536*H floats (dynamic LDS)
  for (int i = threadIdx.x; i < DF * H; i += blockDim.x) sW[i] = W[i];
  __syncthreads();
  int node = blockIdx.x * (blockDim.x >> 5) + (threadIdx.x >> 5);
  if (node >= N_NODES) return;
  int lane = threadIdx.x & 31;
  float acc[H];
#pragma unroll
  for (int hh = 0; hh < H; ++hh) acc[hh] = 0.0f;
  const float* xr = dense + (long)node * DF;
  for (int k = lane; k < DF; k += 32) {
    float v = xr[k];
#pragma unroll
    for (int hh = 0; hh < H; ++hh) acc[hh] += v * sW[k * H + hh];
  }
#pragma unroll
  for (int hh = 0; hh < H; ++hh) {
    float a = acc[hh];
    for (int off = 16; off; off >>= 1) a += __shfl_xor(a, off, 32);
    if (lane == 0) hout[(long)node * H + hh] = a;
  }
}
template <int H>
__global__ void gat_pass1(const int* __restrict__ src, const int* __restrict__ dst,
                          const float* __restrict__ hb, const float* __restrict__ asrc,
                          const float* __restrict__ adst, unsigned* __restrict__ mmax) {
  int i = blockIdx.x * blockDim.x + threadIdx.x;
  if (i >= N_EDGES + N_NODES) return;
  int s, t;
  if (i < N_EDGES) { s = src[i]; t = dst[i]; } else { s = t = i - N_EDGES; }
#pragma unroll
  for (int hh = 0; hh < H; ++hh) {
    float e = hb[(long)s * H + hh] * asrc[hh] + hb[(long)t * H + hh] * adst[hh];
    e = (e < 0.0f) ? NEG_SLOPE * e : e;
    atomicMax(mmax + (long)t * H + hh, f32_ord(e));
  }
}
template <int H>
__global__ void gat_pass2(const int* __restrict__ src, const int* __restrict__ dst,
                          const float* __restrict__ hb, const float* __restrict__ asrc,
                          const float* __restrict__ adst, const unsigned* __restrict__ mmax,
                          float* __restrict__ den, float* __restrict__ num) {
  int i = blockIdx.x * blockDim.x + threadIdx.x;
  if (i >= N_EDGES + N_NODES) return;
  int s, t;
  if (i < N_EDGES) { s = src[i]; t = dst[i]; } else { s = t = i - N_EDGES; }
#pragma unroll
  for (int hh = 0; hh < H; ++hh) {
    float e = hb[(long)s * H + hh] * asrc[hh] + hb[(long)t * H + hh] * adst[hh];
    e = (e < 0.0f) ? NEG_SLOPE * e : e;
    float ex = __expf(e - ord_f32(mmax[(long)t * H + hh]));
    atomicAdd(den + (long)t * H + hh, ex);
    atomicAdd(num + (long)t * H + hh, ex * hb[(long)s * H + hh]);
  }
}
__global__ void gat_final(const float* __restrict__ num1, const float* __restrict__ den1,
                          const float* __restrict__ num2, const float* __restrict__ den2,
                          const float* __restrict__ b1, const float* __restrict__ b2,
                          float* __restrict__ out) {
  int n = blockIdx.x * blockDim.x + threadIdx.x;
  if (n >= N_NODES) return;
  float a1 = 0.0f, a2 = 0.0f;
#pragma unroll
  for (int hh = 0; hh < 4;  ++hh) a1 += num1[(long)n * 4 + hh]  / den1[(long)n * 4 + hh];
#pragma unroll
  for (int hh = 0; hh < 16; ++hh) a2 += num2[(long)n * 16 + hh] / den2[(long)n * 16 + hh];
  a1 = a1 * 0.25f    + b1[0];
  a2 = a2 * 0.0625f  + b2[0];
  out[n] = 0.5f * (a1 + a2);
}

// ---------- host ----------
static inline int nblk(long n) { return (int)((n + 255) / 256); }

extern "C" void kernel_launch(void* const* d_in, const int* in_sizes, int n_in,
                              void* d_out, int out_size, void* d_ws, size_t ws_size,
                              hipStream_t stream) {
  (void)in_sizes; (void)n_in; (void)out_size; (void)ws_size;
  const float* x    = (const float*)d_in[0];
  const int*   ei   = (const int*)  d_in[1];
  const float* ew   = (const float*)d_in[2];
  const float* W1   = (const float*)d_in[3];
  const float* b1   = (const float*)d_in[4];
  const float* W2   = (const float*)d_in[5];
  const float* b2   = (const float*)d_in[6];
  const float* Wl   = (const float*)d_in[7];
  const float* Wr   = (const float*)d_in[8];
  const float* sb   = (const float*)d_in[9];
  const float* g1W  = (const float*)d_in[10];
  const float* g1as = (const float*)d_in[11];
  const float* g1ad = (const float*)d_in[12];
  const float* g1b  = (const float*)d_in[13];
  const float* g2W  = (const float*)d_in[14];
  const float* g2as = (const float*)d_in[15];
  const float* g2ad = (const float*)d_in[16];
  const float* g2b  = (const float*)d_in[17];
  const int* src = ei;
  const int* dst = ei + N_EDGES;

  float* xout  = (float*)d_out;            // [N]
  float* dense = (float*)d_out + N_NODES;  // [N,1536] strided region

  // workspace carving (256B aligned)
  char* wp = (char*)d_ws;
  auto carve = [&](size_t bytes) { void* p = (void*)wp; wp += (bytes + 255) & ~(size_t)255; return p; };
  float*          deg   = (float*)carve((size_t)N_NODES * 4);
  float*          dinv  = (float*)carve((size_t)N_NODES * 4);
  float*          cnt   = (float*)carve((size_t)N_NODES * 4);
  float*          normw = (float*)carve((size_t)N_EDGES * 4);
  unsigned short* xb    = (unsigned short*)carve((size_t)N_NODES * KP1 * 2);
  unsigned short* W1t   = (unsigned short*)carve((size_t)HID * KP1 * 2);
  unsigned short* W2t   = (unsigned short*)carve((size_t)HID * HID * 2);
  unsigned short* Wlt   = (unsigned short*)carve((size_t)HID * HID * 2);
  unsigned short* Wrt   = (unsigned short*)carve((size_t)HID * HID * 2);
  float*          hbuf  = (float*)carve((size_t)N_NODES * HID * 4);  // h1 -> h2 -> x3pre
  float*          agg   = (float*)carve((size_t)N_NODES * HID * 4);  // agg1 -> agg2 -> sums
  unsigned short* x1b   = (unsigned short*)carve((size_t)N_NODES * HID * 2);
  unsigned short* mnb   = (unsigned short*)carve((size_t)N_NODES * HID * 2);
  float*          gh1   = (float*)carve((size_t)N_NODES * 4 * 4);
  float*          gh2   = (float*)carve((size_t)N_NODES * 16 * 4);
  unsigned*       mm1   = (unsigned*)carve((size_t)N_NODES * 4 * 4);
  float*          dn1   = (float*)carve((size_t)N_NODES * 4 * 4);
  float*          nm1   = (float*)carve((size_t)N_NODES * 4 * 4);
  unsigned*       mm2   = (unsigned*)carve((size_t)N_NODES * 16 * 4);
  float*          dn2   = (float*)carve((size_t)N_NODES * 16 * 4);
  float*          nm2   = (float*)carve((size_t)N_NODES * 16 * 4);

  const long NH  = (long)N_NODES * HID;
  const long EC  = (long)N_EDGES * (HID / 4);
  const int  EN  = N_EDGES + N_NODES;
  const int  gemmBlocks = ((N_NODES / 16) * (HID / 16) + 7) / 8;   // 8 waves/block

  // --- normalization constants ---
  fill_u32<<<nblk(N_NODES), 256, 0, stream>>>((unsigned*)deg, 0u, N_NODES);
  deg_k  <<<nblk(N_EDGES), 256, 0, stream>>>(dst, ew, deg);
  dinv_k <<<nblk(N_NODES), 256, 0, stream>>>(deg, dinv);
  norm_k <<<nblk(N_EDGES), 256, 0, stream>>>(src, dst, ew, dinv, normw);

  // --- bf16 conversions ---
  conv_x <<<nblk((long)N_NODES * KP1), 256, 0, stream>>>(x, xb);
  conv_wt<<<nblk((long)HID * KP1), 256, 0, stream>>>(W1, W1t, F_IN, KP1, HID);
  conv_wt<<<nblk((long)HID * HID), 256, 0, stream>>>(W2, W2t, HID, HID, HID);
  conv_wt<<<nblk((long)HID * HID), 256, 0, stream>>>(Wl, Wlt, HID, HID, HID);
  conv_wt<<<nblk((long)HID * HID), 256, 0, stream>>>(Wr, Wrt, HID, HID, HID);

  // --- GCN layer 1: h1 = x@W1 ; aggregate ; epilogue -> x1 (dense off 0, bf16 copy) ---
  gemm_bf16_wmma<<<gemmBlocks, 256, 0, stream>>>(xb, W1t, hbuf, N_NODES, HID, KP1, 0);
  fill_u32<<<nblk(NH), 256, 0, stream>>>((unsigned*)agg, 0u, NH);
  gcn_agg <<<nblk(EC), 256, 0, stream>>>(src, dst, normw, hbuf, agg);
  gcn_epi <<<nblk(NH), 256, 0, stream>>>(agg, hbuf, dinv, b1, x1b, dense, 0);

  // --- GCN layer 2: h2 = x1@W2 ; aggregate ; epilogue -> x2 (dense off 512) ---
  gemm_bf16_wmma<<<gemmBlocks, 256, 0, stream>>>(x1b, W2t, hbuf, N_NODES, HID, HID, 0);
  fill_u32<<<nblk(NH), 256, 0, stream>>>((unsigned*)agg, 0u, NH);
  gcn_agg <<<nblk(EC), 256, 0, stream>>>(src, dst, normw, hbuf, agg);
  gcn_epi <<<nblk(NH), 256, 0, stream>>>(agg, hbuf, dinv, b2, (unsigned short*)nullptr, dense, HID);

  // --- SAGE: mean-aggregate x1 ; x3 = mean@Wl + x1@Wr + b -> dense off 1024 ---
  fill_u32<<<nblk(NH), 256, 0, stream>>>((unsigned*)agg, 0u, NH);
  fill_u32<<<nblk(N_NODES), 256, 0, stream>>>((unsigned*)cnt, 0u, N_NODES);
  cnt_k   <<<nblk(N_EDGES), 256, 0, stream>>>(dst, cnt);
  sage_agg<<<nblk(EC), 256, 0, stream>>>(src, dst, dense, agg);
  mean_k  <<<nblk(NH), 256, 0, stream>>>(agg, cnt, mnb);
  gemm_bf16_wmma<<<gemmBlocks, 256, 0, stream>>>(mnb, Wlt, hbuf, N_NODES, HID, HID, 0);
  gemm_bf16_wmma<<<gemmBlocks, 256, 0, stream>>>(x1b, Wrt, hbuf, N_NODES, HID, HID, 1);
  sage_epi<<<nblk(NH), 256, 0, stream>>>(hbuf, sb, dense);

  // --- GAT projections (wave per node, weights in dynamic LDS) ---
  gat_proj<4> <<<(N_NODES + 7) / 8, 256, DF * 4  * sizeof(float), stream>>>(dense, g1W, gh1);
  gat_proj<16><<<(N_NODES + 7) / 8, 256, DF * 16 * sizeof(float), stream>>>(dense, g2W, gh2);

  // --- GAT attention: segment max -> fused exp-sum / weighted-sum -> combine ---
  fill_u32<<<nblk((long)N_NODES * 4),  256, 0, stream>>>(mm1, 0u, (long)N_NODES * 4);
  fill_u32<<<nblk((long)N_NODES * 4),  256, 0, stream>>>((unsigned*)dn1, 0u, (long)N_NODES * 4);
  fill_u32<<<nblk((long)N_NODES * 4),  256, 0, stream>>>((unsigned*)nm1, 0u, (long)N_NODES * 4);
  fill_u32<<<nblk((long)N_NODES * 16), 256, 0, stream>>>(mm2, 0u, (long)N_NODES * 16);
  fill_u32<<<nblk((long)N_NODES * 16), 256, 0, stream>>>((unsigned*)dn2, 0u, (long)N_NODES * 16);
  fill_u32<<<nblk((long)N_NODES * 16), 256, 0, stream>>>((unsigned*)nm2, 0u, (long)N_NODES * 16);
  gat_pass1<4> <<<nblk(EN), 256, 0, stream>>>(src, dst, gh1, g1as, g1ad, mm1);
  gat_pass1<16><<<nblk(EN), 256, 0, stream>>>(src, dst, gh2, g2as, g2ad, mm2);
  gat_pass2<4> <<<nblk(EN), 256, 0, stream>>>(src, dst, gh1, g1as, g1ad, mm1, dn1, nm1);
  gat_pass2<16><<<nblk(EN), 256, 0, stream>>>(src, dst, gh2, g2as, g2ad, mm2, dn2, nm2);
  gat_final<<<nblk(N_NODES), 256, 0, stream>>>(nm1, dn1, nm2, dn2, g1b, g2b, xout);
}
